// EireneGNN_4939212390552
// MI455X (gfx1250) — matrix operand order repaired and checked by reading, not compiled, not run on hardware
//
#include <hip/hip_runtime.h>
#include <hip/hip_bf16.h>

#define NN 20000
#define EE 320000
#define HH 128
#define LL 6
#define MSG_K 288            // 259 padded to 9 K-tiles of 32
#define MIN_STRIDE 296       // LDS row stride (halves) for m_in tile, bank-friendly
#define HID_STRIDE 136       // LDS row stride (halves) for hidden tile

typedef __attribute__((ext_vector_type(16))) __bf16 v16bf;
typedef __attribute__((ext_vector_type(8)))  float  v8f;

union Frag {
    uint4 u[2];
    v16bf v;
};

// Copy 16 bytes global -> LDS via GLOBAL_LOAD_ASYNC_TO_LDS_B128 (ASYNCcnt-tracked,
// no VGPR round trip). vdst = 32-bit LDS byte offset (low 32 bits of a generic
// LDS pointer, ISA 10.2 truncation rule); vaddr = 64-bit global address, GV mode.
__device__ __forceinline__ void g2l_b128(unsigned short* lds, const unsigned short* g) {
    unsigned int       loff  = (unsigned int)(uintptr_t)lds;
    unsigned long long gaddr = (unsigned long long)(uintptr_t)g;
    asm volatile("global_load_async_to_lds_b128 %0, %1, off"
                 :
                 : "v"(loff), "v"(gaddr)
                 : "memory");
}

__device__ __forceinline__ void gather_fence() {
    asm volatile("s_wait_asynccnt 0x0" ::: "memory");
}

__device__ __forceinline__ unsigned short f2bf(float f) {
    unsigned int u = __float_as_uint(f);
    unsigned int r = u + 0x7fffu + ((u >> 16) & 1u);   // round to nearest even
    return (unsigned short)(r >> 16);
}

__device__ __forceinline__ float silu(float x) {
    return x / (1.0f + __expf(-x));
}

// ---------------- utility kernels ----------------

__global__ void k_zero(float* __restrict__ agg, float* __restrict__ deg) {
    int idx = blockIdx.x * blockDim.x + threadIdx.x;
    if (idx < NN * HH) agg[idx] = 0.0f;
    if (idx < NN) deg[idx] = 0.0f;
}

__global__ void k_deg(const int* __restrict__ ei, float* __restrict__ deg) {
    int e = blockIdx.x * blockDim.x + threadIdx.x;
    if (e < EE) atomicAdd(&deg[ei[EE + e]], 1.0f);
}

// Swizzle conv_w1 (L,259,128) f32 -> bf16 B-fragments: [l][kt(9)][nt(8)][lane(32)][i(16)]
__global__ void k_swz1(const float* __restrict__ w1, unsigned short* __restrict__ w1s) {
    int idx = blockIdx.x * blockDim.x + threadIdx.x;
    const int total = LL * 9 * 8 * 512;
    if (idx >= total) return;
    int l    = idx / (9 * 8 * 512);
    int rem  = idx % (9 * 8 * 512);
    int kt   = rem / (8 * 512);
    int rem2 = rem % (8 * 512);
    int nt   = rem2 >> 9;
    int lr   = rem2 & 511;
    int lane = lr >> 4;
    int i    = lr & 15;
    int k = kt * 32 + ((i >> 3) << 4) + ((lane >> 4) << 3) + (i & 7);   // B layout: K = (i/8)*16 + (lane/16)*8 + i%8
    int n = nt * 16 + (lane & 15);
    float v = (k < 259) ? w1[((size_t)l * 259 + k) * HH + n] : 0.0f;
    w1s[idx] = f2bf(v);
}

// Swizzle conv_w2 (L,128,128) -> [l][kt(4)][nt(8)][lane][i]
__global__ void k_swz2(const float* __restrict__ w2, unsigned short* __restrict__ w2s) {
    int idx = blockIdx.x * blockDim.x + threadIdx.x;
    const int total = LL * 4 * 8 * 512;
    if (idx >= total) return;
    int l    = idx / (4 * 8 * 512);
    int rem  = idx % (4 * 8 * 512);
    int kt   = rem / (8 * 512);
    int rem2 = rem % (8 * 512);
    int nt   = rem2 >> 9;
    int lr   = rem2 & 511;
    int lane = lr >> 4;
    int i    = lr & 15;
    int k = kt * 32 + ((i >> 3) << 4) + ((lane >> 4) << 3) + (i & 7);
    int n = nt * 16 + (lane & 15);
    w2s[idx] = f2bf(w2[((size_t)l * HH + k) * HH + n]);
}

// Generic dense: out[r][j] = act(sum_k in[r][k]*w[k][j] + b[j])
__global__ void k_dense(const float* __restrict__ in, const float* __restrict__ w,
                        const float* __restrict__ b, float* __restrict__ out,
                        int rows, int K, int Nout, int act) {
    int idx = blockIdx.x * blockDim.x + threadIdx.x;
    if (idx >= rows * Nout) return;
    int r = idx / Nout, j = idx % Nout;
    float s = b[j];
    const float* ip = in + (size_t)r * K;
    for (int k = 0; k < K; ++k) s += ip[k] * w[(size_t)k * Nout + j];
    if (act) s = silu(s);
    out[idx] = s;
}

// Encoder stage 2: writes f32 h and bf16 mirror
__global__ void k_enc2(const float* __restrict__ in, const float* __restrict__ w,
                       const float* __restrict__ b, float* __restrict__ h,
                       unsigned short* __restrict__ hbf) {
    int idx = blockIdx.x * blockDim.x + threadIdx.x;
    if (idx >= NN * HH) return;
    int r = idx >> 7, j = idx & 127;
    float s = b[j];
    const float* ip = in + ((size_t)r << 7);
    for (int k = 0; k < HH; ++k) s += ip[k] * w[(k << 7) + j];
    h[idx] = s;
    hbf[idx] = f2bf(s);
}

// Node update: h += agg/deg; refresh bf16 mirror; re-zero agg for next layer
__global__ void k_update(float* __restrict__ h, float* __restrict__ agg,
                         const float* __restrict__ deg, unsigned short* __restrict__ hbf) {
    int idx = blockIdx.x * blockDim.x + threadIdx.x;
    if (idx >= NN * HH) return;
    int n = idx >> 7;
    float d = deg[n];
    if (d < 1.0f) d = 1.0f;
    float v = h[idx] + agg[idx] / d;
    h[idx] = v;
    hbf[idx] = f2bf(v);
    agg[idx] = 0.0f;
}

// ---------------- main edge-MLP WMMA kernel ----------------
// 64 edges per block, 8 waves. Wave w: mtile = w>>1 (16 edges), nhalf = w&1 (4 N-tiles).
__global__ __launch_bounds__(256) void k_edge(
    const unsigned short* __restrict__ hbf, const int* __restrict__ ei,
    const float* __restrict__ eattr,
    const unsigned short* __restrict__ w1s, const float* __restrict__ b1,
    const unsigned short* __restrict__ w2s, const float* __restrict__ b2,
    float* __restrict__ agg) {
    __shared__ unsigned short sh_min[64 * MIN_STRIDE]; // 64 x 288 (padded) bf16 message inputs
    __shared__ unsigned short sh_hid[64 * HID_STRIDE]; // 64 x 128 (padded) bf16 hidden
    __shared__ int sh_dst[64];

    const int tid = threadIdx.x;
    const int e0  = blockIdx.x * 64;

    // Cooperative gather: each edge row = [h_bf16[dst] (128) | h_bf16[src] (128) | attr | 0]
    // Async global->LDS copies (ASYNCcnt) avoid the VGPR round trip.
    for (int idx = tid; idx < 64 * 16; idx += 256) {
        int el = idx >> 4, seg = idx & 15;
        int e  = e0 + el;
        int src = ei[e];
        int dst = ei[EE + e];
        unsigned short* row = sh_min + el * MIN_STRIDE;
        g2l_b128(row + seg * 8,       hbf + ((size_t)dst << 7) + seg * 8); // cols   0..127
        g2l_b128(row + 128 + seg * 8, hbf + ((size_t)src << 7) + seg * 8); // cols 128..255
        if (seg == 0) sh_dst[el] = dst;
    }
    if (tid < 64) {
        unsigned short* row = sh_min + tid * MIN_STRIDE;
        const float* ea = eattr + (size_t)(e0 + tid) * 3;
        row[256] = f2bf(ea[0]);
        row[257] = f2bf(ea[1]);
        row[258] = f2bf(ea[2]);
        for (int c = 259; c < MIN_STRIDE; ++c) row[c] = 0;  // zero K padding
    }
    gather_fence();     // s_wait_asynccnt 0 (this wave's async copies landed)
    __syncthreads();

    const int lane  = tid & 31;
    const int wv    = tid >> 5;
    const int mt    = wv >> 1;
    const int nh    = (wv & 1) * 4;
    const int row   = mt * 16 + (lane & 15);       // A-frag row (edge within block)
    const int colA0 = (lane >> 4) * 8;             // A layout: lanes 16..31 offset by 8 in K

    // ---- GEMM1: (16 x 288) @ (288 x 128), bias in accumulator ----
    v8f acc[4];
    for (int t = 0; t < 4; ++t) {
        float bv = b1[(nh + t) * 16 + (lane & 15)];
        for (int j = 0; j < 8; ++j) acc[t][j] = bv;
    }
    for (int kt = 0; kt < 9; ++kt) {
        Frag a;
        const uint4* ap = (const uint4*)(sh_min + row * MIN_STRIDE + kt * 32 + colA0);
        a.u[0] = ap[0];      // K = kt*32 + colA0 + 0..7
        a.u[1] = ap[2];      // K = kt*32 + colA0 + 16..23
        #pragma unroll
        for (int t = 0; t < 4; ++t) {
            int nt = nh + t;
            Frag b;
            const uint4* bp = (const uint4*)(w1s + (((size_t)kt * 8 + nt) << 9) + (lane << 4));
            b.u[0] = bp[0];
            b.u[1] = bp[1];
            acc[t] = __builtin_amdgcn_wmma_f32_16x16x32_bf16(
                false, a.v, false, b.v, (short)0, acc[t], false, false);
        }
    }
    // SiLU -> bf16 hidden tile in LDS
    for (int t = 0; t < 4; ++t) {
        int n = (nh + t) * 16 + (lane & 15);
        #pragma unroll
        for (int j = 0; j < 8; ++j) {
            int m = mt * 16 + (lane >> 4) * 8 + j;
            sh_hid[m * HID_STRIDE + n] = f2bf(silu(acc[t][j]));
        }
    }
    __syncthreads();

    // ---- GEMM2: (16 x 128) @ (128 x 128) ----
    v8f acc2[4];
    for (int t = 0; t < 4; ++t) {
        float bv = b2[(nh + t) * 16 + (lane & 15)];
        for (int j = 0; j < 8; ++j) acc2[t][j] = bv;
    }
    for (int kt = 0; kt < 4; ++kt) {
        Frag a;
        const uint4* ap = (const uint4*)(sh_hid + row * HID_STRIDE + kt * 32 + colA0);
        a.u[0] = ap[0];
        a.u[1] = ap[2];
        #pragma unroll
        for (int t = 0; t < 4; ++t) {
            int nt = nh + t;
            Frag b;
            const uint4* bp = (const uint4*)(w2s + (((size_t)kt * 8 + nt) << 9) + (lane << 4));
            b.u[0] = bp[0];
            b.u[1] = bp[1];
            acc2[t] = __builtin_amdgcn_wmma_f32_16x16x32_bf16(
                false, a.v, false, b.v, (short)0, acc2[t], false, false);
        }
    }
    // Scatter-add messages into agg[dst]
    for (int t = 0; t < 4; ++t) {
        int n = (nh + t) * 16 + (lane & 15);
        #pragma unroll
        for (int j = 0; j < 8; ++j) {
            int m = mt * 16 + (lane >> 4) * 8 + j;
            atomicAdd(&agg[((size_t)sh_dst[m] << 7) + n], acc2[t][j]);
        }
    }
}

// ---------------- host launch ----------------

extern "C" void kernel_launch(void* const* d_in, const int* in_sizes, int n_in,
                              void* d_out, int out_size, void* d_ws, size_t ws_size,
                              hipStream_t stream) {
    const float* x       = (const float*)d_in[0];
    const int*   ei      = (const int*)  d_in[1];   // (2,E) int32
    const float* eattr   = (const float*)d_in[2];
    const float* enc_w1  = (const float*)d_in[3];
    const float* enc_b1  = (const float*)d_in[4];
    const float* enc_w2  = (const float*)d_in[5];
    const float* enc_b2  = (const float*)d_in[6];
    const float* conv_w1 = (const float*)d_in[7];
    const float* conv_b1 = (const float*)d_in[8];
    const float* conv_w2 = (const float*)d_in[9];
    const float* conv_b2 = (const float*)d_in[10];
    const float* dec_w1  = (const float*)d_in[11];
    const float* dec_b1  = (const float*)d_in[12];
    const float* dec_w2  = (const float*)d_in[13];
    const float* dec_b2  = (const float*)d_in[14];
    const float* dec_w3  = (const float*)d_in[15];
    const float* dec_b3  = (const float*)d_in[16];
    float* out = (float*)d_out;

    char* ws = (char*)d_ws;
    size_t off = 0;
    float*          h    = (float*)(ws + off);          off += (size_t)NN * HH * 4;  // 10.24 MB
    unsigned short* hbf  = (unsigned short*)(ws + off); off += (size_t)NN * HH * 2;  //  5.12 MB
    float*          agg  = (float*)(ws + off);          off += (size_t)NN * HH * 4;  // 10.24 MB (also enc tmp / dec tmp)
    float*          deg  = (float*)(ws + off);          off += (size_t)NN * 4 + 192; //  deg, padded
    float*          tmp2 = (float*)(ws + off);          off += (size_t)NN * 64 * 4;  //  5.12 MB
    unsigned short* w1s  = (unsigned short*)(ws + off); off += (size_t)LL * 9 * 8 * 512 * 2;
    unsigned short* w2s  = (unsigned short*)(ws + off); off += (size_t)LL * 4 * 8 * 512 * 2;

    const int B = 256;
    const int gNH = (NN * HH + B - 1) / B;

    // Weight swizzle (bf16 B-fragment order)
    k_swz1<<<(LL * 9 * 8 * 512 + B - 1) / B, B, 0, stream>>>(conv_w1, w1s);
    k_swz2<<<(LL * 4 * 8 * 512 + B - 1) / B, B, 0, stream>>>(conv_w2, w2s);

    // Encoder: x -> silu(x@W1+b1) -> @W2+b2 (agg buffer reused as tmp here)
    k_dense<<<gNH, B, 0, stream>>>(x, enc_w1, enc_b1, agg, NN, 14, HH, 1);
    k_enc2<<<gNH, B, 0, stream>>>(agg, enc_w2, enc_b2, h, hbf);

    // Degree + zero agg (after encoder tmp use)
    k_zero<<<gNH, B, 0, stream>>>(agg, deg);
    k_deg<<<(EE + B - 1) / B, B, 0, stream>>>(ei, deg);

    // Message-passing layers
    for (int l = 0; l < LL; ++l) {
        k_edge<<<EE / 64, B, 0, stream>>>(
            hbf, ei, eattr,
            w1s + (size_t)l * 9 * 8 * 512, conv_b1 + (size_t)l * HH,
            w2s + (size_t)l * 4 * 8 * 512, conv_b2 + (size_t)l * HH,
            agg);
        k_update<<<gNH, B, 0, stream>>>(h, agg, deg, hbf);
    }

    // Decoder (agg buffer reused as tmp)
    k_dense<<<gNH, B, 0, stream>>>(h, dec_w1, dec_b1, agg, NN, HH, HH, 1);
    k_dense<<<(NN * 64 + B - 1) / B, B, 0, stream>>>(agg, dec_w2, dec_b2, tmp2, NN, HH, 64, 1);
    k_dense<<<(NN * 9 + B - 1) / B, B, 0, stream>>>(tmp2, dec_w3, dec_b3, out, NN, 64, 9, 0);
}
